// RNN_695784702463
// MI455X (gfx1250) — compile-verified, matrix operand order checked by reading, not run on hardware
//
#include <hip/hip_runtime.h>
#include <cmath>

#define T_STEPS 512
#define BATCH   64
#define DIM     1024
#define HID     1024

typedef __attribute__((ext_vector_type(16))) _Float16 v16h;
typedef __attribute__((ext_vector_type(8)))  _Float16 v8h;
typedef __attribute__((ext_vector_type(8)))  float    v8f;
typedef __attribute__((ext_vector_type(4)))  unsigned ui4;
typedef __attribute__((ext_vector_type(8)))  int      i8v;
typedef __attribute__((ext_vector_type(4)))  int      i4v;

#if defined(__has_builtin)
#if __has_builtin(__builtin_amdgcn_tensor_load_to_lds)
#define USE_TDM 1
#endif
#endif
#ifndef USE_TDM
#define USE_TDM 0
#endif

union HV { v16h v; v8h h[2]; };

// Load one 16x32 f16 fragment from a row-major tile (ld = leading dim), per
// CDNA5 16-bit WMMA striping: lane<16 -> row=lane, K in [0,8)+[16,24);
// lane>=16 -> row=lane-16, K in [8,16)+[24,32).  Two 16B loads per lane.
__device__ __forceinline__ v16h load_frag(const _Float16* __restrict__ tile,
                                          int ld, int lane) {
  const int r   = lane & 15;
  const int sel = lane >> 4;
  const _Float16* p = tile + (size_t)r * ld + sel * 8;
  HV u;
  u.h[0] = *(const v8h*)(p);
  u.h[1] = *(const v8h*)(p + 16);
  return u.v;
}

#if USE_TDM
// One 1-D TDM transfer: n_qwords 8-byte elements, global -> LDS.
// D# group0: count=1, lds_addr[63:32], global_addr[120:64], type=2[127:126].
// D# group1: data_size=3 (8B) [17:16], tensor_dim0 [79:48], tensor_dim1=1
// [111:80], tile_dim0 [127:112], tensor_dim0_stride [207:160].
// This toolchain's builtin takes 6 args:
//   (uint32x4 g0, int32x8 g1, int32x4 g2, int32x4 g3, int32x8 x, i32 cpol)
__device__ __forceinline__ void tdm_load_1d(unsigned lds_addr,
                                            const void* gsrc,
                                            unsigned n_qwords) {
  const unsigned long long ga = (unsigned long long)(size_t)gsrc;
  ui4 g0;
  g0.x = 1u;                                   // count=1, user mode
  g0.y = lds_addr;                             // LDS byte address
  g0.z = (unsigned)ga;                         // global_addr[31:0]
  g0.w = (unsigned)(ga >> 32) | (2u << 30);    // global_addr[56:32], type=2
  i8v g1;
  g1[0] = (int)(3u << 16);                         // data_size = 8 bytes
  g1[1] = (int)((n_qwords & 0xffffu) << 16);       // tensor_dim0 lo16
  g1[2] = (int)(((n_qwords >> 16) & 0xffffu) | (1u << 16)); // dim0 hi16, dim1=1
  g1[3] = (int)((n_qwords & 0xffffu) << 16);       // tile_dim0
  g1[4] = 0;                                        // tile_dim1/2 unused
  g1[5] = (int)n_qwords;                            // tensor_dim0_stride lo32
  g1[6] = 0;
  g1[7] = 0;
  const i4v z4 = {0, 0, 0, 0};
  const i8v z8 = {0, 0, 0, 0, 0, 0, 0, 0};
  __builtin_amdgcn_tensor_load_to_lds(g0, g1, z4, z4, z8, 0);
}
#endif

// ---------------- data staging kernels ----------------

__global__ void k_f32_to_f16(const float* __restrict__ in,
                             _Float16* __restrict__ out, int n) {
  for (int i = blockIdx.x * blockDim.x + threadIdx.x; i < n;
       i += gridDim.x * blockDim.x)
    out[i] = (_Float16)in[i];
}

// in: [R][C] f32 row-major  ->  out: [C][R] f16 row-major (transpose)
__global__ void k_transpose_f16(const float* __restrict__ in,
                                _Float16* __restrict__ out, int R, int C) {
  int total = R * C;
  for (int i = blockIdx.x * blockDim.x + threadIdx.x; i < total;
       i += gridDim.x * blockDim.x) {
    int c = i / R;
    int r = i % R;
    out[i] = (_Float16)in[(size_t)r * C + c];
  }
}

__global__ void k_bias(const float* __restrict__ a, const float* __restrict__ b,
                       float* __restrict__ o, int n) {
  for (int i = blockIdx.x * blockDim.x + threadIdx.x; i < n;
       i += gridDim.x * blockDim.x)
    o[i] = a[i] + b[i];
}

// ---------------- phase 1: U[t,b,:] = x_t @ Wih + (bih+bhh) ----------------
// Block = 256 threads (8 waves).  Wave w owns M-rows [bx*256 + w*32, +32)
// (2 M-tiles) x N-cols [by*64, +64) (4 N-tiles) = 8 accumulators.
// Per K-step: 2 A-loads + 4 B-loads feed 8 WMMAs (every B reused twice).
__global__ void k_gemm_xWih(const _Float16* __restrict__ X,
                            const _Float16* __restrict__ WT,
                            const float* __restrict__ bias,
                            float* __restrict__ U) {
  const int lane = threadIdx.x & 31;
  const int wave = threadIdx.x >> 5;
  const int m0 = blockIdx.x * 256 + wave * 32;
  const int n0 = blockIdx.y * 64;

  v8f acc[2][4] = {};
  const _Float16* xrow0 = X + (size_t)m0 * DIM;
  const _Float16* xrow1 = X + (size_t)(m0 + 16) * DIM;

  for (int k0 = 0; k0 < DIM; k0 += 32) {
    __builtin_prefetch(xrow0 + k0 + 256, 0, 1);  // global_prefetch_b8
    __builtin_prefetch(xrow1 + k0 + 256, 0, 1);
    v16h a0 = load_frag(xrow0 + k0, DIM, lane);
    v16h a1 = load_frag(xrow1 + k0, DIM, lane);
#pragma unroll
    for (int j = 0; j < 4; ++j) {
      v16h b = load_frag(WT + (size_t)(n0 + j * 16) * DIM + k0, DIM, lane);
      acc[0][j] = __builtin_amdgcn_wmma_f32_16x16x32_f16(
          false, a0, false, b, (short)0, acc[0][j], false, false);
      acc[1][j] = __builtin_amdgcn_wmma_f32_16x16x32_f16(
          false, a1, false, b, (short)0, acc[1][j], false, false);
    }
  }

  const int sel = lane >> 4;
  const int cb  = lane & 15;
#pragma unroll
  for (int s = 0; s < 2; ++s) {
#pragma unroll
    for (int j = 0; j < 4; ++j) {
      const int col = n0 + j * 16 + cb;
      const float bb = bias[col];
#pragma unroll
      for (int i = 0; i < 8; ++i) {
        const int row = m0 + s * 16 + sel * 8 + i;
        U[(size_t)row * HID + col] = acc[s][j][i] + bb;
      }
    }
  }
}

// ---------------- phase 2: persistent recurrence ----------------
// 16 blocks x 256 threads.  Block keeps a 64-col slice of Whh^T resident in
// LDS (128KB) for all 512 steps; h_{t-1} (128KB f16) is pulled into LDS each
// step by a single TDM tensor_load_to_lds DMA.  Steps separated by an
// atomic-counter grid barrier.
__global__ void k_rnn_steps(const _Float16* __restrict__ WhhT,
                            _Float16* __restrict__ hbuf,  // [2][B*H] f16
                            float* __restrict__ out,      // [T][B*H] f32
                            unsigned* __restrict__ cnt,   // [T]
                            int nblocks) {
  extern __shared__ __align__(16) _Float16 smem[];
  _Float16* sW = smem;              // [64][HID]  Whh^T slice (cols n0..n0+63)
  _Float16* sH = smem + 64 * HID;   // [64][HID]  h_{t-1}

  const int tid  = threadIdx.x;
  const int lane = tid & 31;
  const int wave = tid >> 5;
  const int n0   = blockIdx.x * 64;

  // one-time weight slice load: 64 contiguous rows of Whh^T = 128KB
#if USE_TDM
  if (wave == 0) {
    tdm_load_1d((unsigned)(size_t)(void*)sW, WhhT + (size_t)n0 * HID,
                64 * HID * 2 / 8);
    __builtin_amdgcn_s_wait_tensorcnt(0);
  }
#else
  {
    const uint4* src = (const uint4*)(WhhT + (size_t)n0 * HID);
    uint4* dst = (uint4*)sW;
    for (int i = tid; i < 64 * HID / 8; i += blockDim.x) dst[i] = src[i];
  }
#endif

  const int mt = (wave & 3) * 16;   // M-tile (batch rows)
  const int nt = (wave >> 2) * 32;  // N-pair base within the 64-col slice

  for (int t = 0; t < T_STEPS; ++t) {
    const int cur = t & 1, nxt = cur ^ 1;
    // broadcast h_{t-1} into LDS
#if USE_TDM
    if (wave == 0) {
      tdm_load_1d((unsigned)(size_t)(void*)sH,
                  hbuf + (size_t)cur * BATCH * HID, BATCH * HID * 2 / 8);
      __builtin_amdgcn_s_wait_tensorcnt(0);
    }
#else
    {
      const uint4* src = (const uint4*)(hbuf + (size_t)cur * BATCH * HID);
      uint4* dst = (uint4*)sH;
      for (int i = tid; i < BATCH * HID / 8; i += blockDim.x) dst[i] = src[i];
    }
#endif
    __syncthreads();

    v8f acc0 = {}, acc1 = {};
    for (int k0 = 0; k0 < HID; k0 += 32) {
      v16h a  = load_frag(sH + (size_t)mt * HID + k0, HID, lane);
      v16h b0 = load_frag(sW + (size_t)nt * HID + k0, HID, lane);
      v16h b1 = load_frag(sW + (size_t)(nt + 16) * HID + k0, HID, lane);
      acc0 = __builtin_amdgcn_wmma_f32_16x16x32_f16(false, a, false, b0,
                                                    (short)0, acc0, false, false);
      acc1 = __builtin_amdgcn_wmma_f32_16x16x32_f16(false, a, false, b1,
                                                    (short)0, acc1, false, false);
    }

    const int sel = lane >> 4, cb = lane & 15;
    float* Ut = out + (size_t)t * BATCH * HID;
    _Float16* hn = hbuf + (size_t)nxt * BATCH * HID;
    const int c0 = n0 + nt + cb, c1 = c0 + 16;
#pragma unroll
    for (int i = 0; i < 8; ++i) {
      const int bm = mt + sel * 8 + i;
      const size_t o0 = (size_t)bm * HID + c0;
      const size_t o1 = (size_t)bm * HID + c1;
      const float h0 = tanhf(Ut[o0] + acc0[i]);
      const float h1 = tanhf(Ut[o1] + acc1[i]);
      Ut[o0] = h0;  Ut[o1] = h1;
      hn[o0] = (_Float16)h0;  hn[o1] = (_Float16)h1;
    }

    __syncthreads();
    __threadfence();
    if (tid == 0) {
      __hip_atomic_fetch_add(&cnt[t], 1u, __ATOMIC_ACQ_REL,
                             __HIP_MEMORY_SCOPE_AGENT);
      while (__hip_atomic_load(&cnt[t], __ATOMIC_ACQUIRE,
                               __HIP_MEMORY_SCOPE_AGENT) < (unsigned)nblocks)
        __builtin_amdgcn_s_sleep(2);
    }
    __syncthreads();
  }
}

// ---------------- launch ----------------

extern "C" void kernel_launch(void* const* d_in, const int* in_sizes, int n_in,
                              void* d_out, int out_size, void* d_ws,
                              size_t ws_size, hipStream_t stream) {
  (void)in_sizes; (void)n_in; (void)out_size; (void)ws_size;
  const float* init = (const float*)d_in[0];  // [B,H]
  const float* xs   = (const float*)d_in[1];  // [T,B,D]
  const float* Wih  = (const float*)d_in[2];  // [D,H]
  const float* bih  = (const float*)d_in[3];  // [H]
  const float* Whh  = (const float*)d_in[4];  // [H,H]
  const float* bhh  = (const float*)d_in[5];  // [H]
  float* out = (float*)d_out;                 // [T,B,H]

  char* ws = (char*)d_ws;
  size_t off = 0;
  auto carve = [&](size_t bytes) -> void* {
    void* p = ws + off;
    off = (off + bytes + 255) & ~(size_t)255;
    return p;
  };
  _Float16* xs16   = (_Float16*)carve((size_t)T_STEPS * BATCH * DIM * 2); // 64MB
  _Float16* WihT16 = (_Float16*)carve((size_t)DIM * HID * 2);             //  2MB
  _Float16* WhhT16 = (_Float16*)carve((size_t)HID * HID * 2);             //  2MB
  float*    bias   = (float*)carve((size_t)HID * 4);
  _Float16* hbuf   = (_Float16*)carve((size_t)2 * BATCH * HID * 2);       // 256KB
  unsigned* cnt    = (unsigned*)carve((size_t)T_STEPS * 4);

  (void)hipMemsetAsync(cnt, 0, T_STEPS * sizeof(unsigned), stream);

  k_f32_to_f16<<<2048, 256, 0, stream>>>(xs, xs16, T_STEPS * BATCH * DIM);
  k_transpose_f16<<<1024, 256, 0, stream>>>(Wih, WihT16, DIM, HID);
  k_transpose_f16<<<1024, 256, 0, stream>>>(Whh, WhhT16, HID, HID);
  k_bias<<<4, 256, 0, stream>>>(bih, bhh, bias, HID);
  k_f32_to_f16<<<256, 256, 0, stream>>>(init, hbuf, BATCH * HID);

  k_gemm_xWih<<<dim3(128, 16), 256, 0, stream>>>(xs16, WihT16, bias, out);

  const int lds_bytes = 2 * 64 * HID * 2;  // 256KB of the 320KB WGP LDS
  (void)hipFuncSetAttribute(reinterpret_cast<const void*>(k_rnn_steps),
                            hipFuncAttributeMaxDynamicSharedMemorySize,
                            lds_bytes);
  k_rnn_steps<<<16, 256, lds_bytes, stream>>>(WhhT16, hbuf, out, cnt, 16);
}
